// DeltaRecurrentUpdate_40218073760069
// MI455X (gfx1250) — matrix operand order repaired and checked by reading, not compiled, not run on hardware
//
#include <hip/hip_runtime.h>
#include <hip/hip_bf16.h>

// Problem constants (match reference)
#define BB 8
#define LL 8192
#define RR 64
#define HH 512
#define EPSN 1e-12f

typedef __attribute__((ext_vector_type(16))) __bf16 bf16x16;
typedef __attribute__((ext_vector_type(8)))  float  f32x8;

union FragBF { bf16x16 v; __bf16 e[16]; uint4 u2[2]; };
union AccF   { f32x8   v; float  e[8]; };

#define WMMA_BF16(A, Bf, Cf) \
    __builtin_amdgcn_wmma_f32_16x16x32_bf16(false, (A), false, (Bf), (short)0, (Cf), false, false)

// ---------------------------------------------------------------------------
// Prep A: transpose+cast weights: WkT/WvT[h*RR + r] = (bf16) W[r*HH + h]
// ---------------------------------------------------------------------------
__global__ __launch_bounds__(256) void prep_weights(
    const float* __restrict__ kw, const float* __restrict__ vw,
    __bf16* __restrict__ WkT, __bf16* __restrict__ WvT)
{
    int i = blockIdx.x * 256 + threadIdx.x;     // i = h*RR + r, i < HH*RR
    if (i >= HH * RR) return;
    int h = i >> 6;          // /RR
    int r = i & (RR - 1);
    WkT[i] = (__bf16)kw[r * HH + h];
    WvT[i] = (__bf16)vw[r * HH + h];
}

// ---------------------------------------------------------------------------
// Prep B: transpose+cast cache: CT[b][d*HH + h] = (bf16) C[b][h*HH + d]
// ---------------------------------------------------------------------------
__global__ __launch_bounds__(256) void prep_cache(
    const float* __restrict__ C, __bf16* __restrict__ CT)
{
    int i = blockIdx.x * 256 + threadIdx.x;     // i < BB*HH*HH
    int b   = i >> 18;                          // /(HH*HH)
    int rem = i & (HH * HH - 1);
    int d   = rem >> 9;
    int h   = rem & (HH - 1);
    CT[i] = (__bf16)C[(size_t)b * HH * HH + (size_t)h * HH + d];
}

// ---------------------------------------------------------------------------
// Kernel 1: per (b, 16-row L-tile): K = normalize(HS@Wk + bk) -> KT (bf16,
// transposed [B,H,L]);  V = HS@Wv + bv - K@C -> VT (bf16, transposed).
// 256 threads = 8 waves; each wave owns 4 h-tiles of 16 columns.
// ---------------------------------------------------------------------------
__global__ __launch_bounds__(256) void k_proj(
    const float* __restrict__ hs,
    const float* __restrict__ key_b, const float* __restrict__ value_b,
    const __bf16* __restrict__ WkT, const __bf16* __restrict__ WvT,
    const __bf16* __restrict__ CT,
    __bf16* __restrict__ KT, __bf16* __restrict__ VT)
{
    __shared__ float  ldsF[16 * HH];   // f32 K tile (pre-normalization)  32 KB
    __shared__ __bf16 ldsK[16 * HH];   // bf16 normalized K tile          16 KB
    __shared__ float  norm2[16];

    const int wg   = blockIdx.x;               // 0 .. B*L/16-1
    const int b    = wg / (LL / 16);
    const int l0   = (wg % (LL / 16)) * 16;
    const int tid  = threadIdx.x;
    const int wave = tid >> 5;
    const int lane = tid & 31;                 // wave32
    const int half = lane >> 4;
    const int col  = lane & 15;

    if (tid < 16) norm2[tid] = 0.f;

    // A-fragments from hidden_states (f32 -> bf16), K-dim = R in 2 steps of 32.
    // A layout: lane m holds K = 16*half..+15 (contiguous).
    const float* hrow = hs + ((size_t)b * LL + l0 + col) * RR;
    FragBF afrag[2];
#pragma unroll
    for (int kk = 0; kk < 2; ++kk) {
        const int base = kk * 32 + half * 16;
#pragma unroll
        for (int i = 0; i < 8; ++i) {
            afrag[kk].e[i]     = (__bf16)hrow[base + i];
            afrag[kk].e[8 + i] = (__bf16)hrow[base + 8 + i];
        }
    }

    // ---- K = HS @ Wk + bk ----
#pragma unroll
    for (int t = 0; t < 4; ++t) {
        const int n0 = (wave * 4 + t) * 16;
        const float bias = key_b[n0 + col];
        AccF acc;
#pragma unroll
        for (int v = 0; v < 8; ++v) acc.e[v] = bias;
#pragma unroll
        for (int kk = 0; kk < 2; ++kk) {
            FragBF bf;
            const __bf16* wrow = WkT + (size_t)(n0 + col) * RR + kk * 32 + half * 16;
            bf.u2[0] = *(const uint4*)(wrow);
            bf.u2[1] = *(const uint4*)(wrow + 8);
            acc.v = WMMA_BF16(afrag[kk].v, bf.v, acc.v);
        }
#pragma unroll
        for (int v = 0; v < 8; ++v)
            ldsF[(v + 8 * half) * HH + n0 + col] = acc.e[v];
    }
    __syncthreads();

    // ---- row L2 norms (f32) ----
    {
        const int row = tid >> 4;              // 0..15
        const int c0  = (tid & 15) * 32;
        float s = 0.f;
#pragma unroll
        for (int i = 0; i < 32; ++i) {
            float x = ldsF[row * HH + c0 + i];
            s += x * x;
        }
        atomicAdd(&norm2[row], s);             // ds_add_f32
    }
    __syncthreads();

    // ---- normalize, quantize, emit KT[b][h][l] ----
    for (int idx = tid; idx < 16 * HH; idx += 256) {
        const int row = idx >> 9;
        const int c   = idx & (HH - 1);
        const float rn = 1.f / fmaxf(sqrtf(norm2[row]), EPSN);
        const __bf16 kb = (__bf16)(ldsF[idx] * rn);
        ldsK[idx] = kb;
        KT[((size_t)b * HH + c) * LL + l0 + row] = kb;
    }
    __syncthreads();

    // ---- V = HS @ Wv + bv - K @ C ----
    const __bf16* ctb = CT + (size_t)b * HH * HH;
#pragma unroll
    for (int t = 0; t < 4; ++t) {
        const int n0 = (wave * 4 + t) * 16;
        const float bias = value_b[n0 + col];
        AccF acc;
#pragma unroll
        for (int v = 0; v < 8; ++v) acc.e[v] = bias;
#pragma unroll
        for (int kk = 0; kk < 2; ++kk) {
            FragBF bf;
            const __bf16* wrow = WvT + (size_t)(n0 + col) * RR + kk * 32 + half * 16;
            bf.u2[0] = *(const uint4*)(wrow);
            bf.u2[1] = *(const uint4*)(wrow + 8);
            acc.v = WMMA_BF16(afrag[kk].v, bf.v, acc.v);
        }
        // kc = K_tile(16xH) @ C(HxH) tile; bf16 WMMA can't negate A/B (ISA:
        // NEG = CNeg only), so accumulate separately and subtract in VALU.
        AccF kc;
#pragma unroll
        for (int v = 0; v < 8; ++v) kc.e[v] = 0.f;
#pragma unroll 4
        for (int kk = 0; kk < 16; ++kk) {
            FragBF ak;
            const __bf16* krow = ldsK + col * HH + kk * 32 + half * 16;
            ak.u2[0] = *(const uint4*)(krow);        // K = base..base+7
            ak.u2[1] = *(const uint4*)(krow + 8);    // K = base+8..base+15
            FragBF bc;
            const __bf16* crow = ctb + (size_t)(n0 + col) * HH + kk * 32 + half * 16;
            bc.u2[0] = *(const uint4*)(crow);
            bc.u2[1] = *(const uint4*)(crow + 8);
            kc.v = WMMA_BF16(ak.v, bc.v, kc.v);
        }
#pragma unroll
        for (int v = 0; v < 8; ++v) {
            const int row = v + 8 * half;
            VT[((size_t)b * HH + n0 + col) * LL + l0 + row] =
                (__bf16)(acc.e[v] - kc.e[v]);
        }
    }
}

// ---------------------------------------------------------------------------
// Kernel 2: C_new = C + K^T @ V. WG = 8 waves covering a 32(h1) x 256(h2)
// output panel; each wave owns a 32x32 tile (4 f32 accumulators). The shared
// 32-row K chunk (32 x 128 l, 8 KB) is double-buffered in LDS via
// global_load_async_to_lds_b128 (ASYNCcnt), A-frags read with ds_load_b128,
// V streams from L2 with prefetch hints. 256 bf16 WMMA k-steps over L=8192.
// ---------------------------------------------------------------------------
__global__ __launch_bounds__(256) void k_outer(
    const __bf16* __restrict__ KT, const __bf16* __restrict__ VT,
    const float* __restrict__ C, float* __restrict__ out)
{
    __shared__ __bf16 ldsA[2][32 * 128];   // 2 x 8 KB K staging buffers

    const int tid  = threadIdx.x;
    const int wave = tid >> 5;
    const int lane = tid & 31;             // wave32
    const int half = lane >> 4;
    const int col  = lane & 15;

    const int bx  = blockIdx.x;            // 256 WGs: b(8) x h1g(16) x h2g(2)
    const int b   = bx >> 5;
    const int rem = bx & 31;
    const int h1  = (rem >> 1) * 32;
    const int h2w = (rem & 1) * 256 + wave * 32;

    // Async staging: thread t copies uint4 {t, t+256}: row = i>>4, seg = i&15
    const int r0 = tid >> 4;               // 0..15
    const int s0 = tid & 15;
    const unsigned long long gaK0 = (unsigned long long)(const char*)
        (KT + ((size_t)b * HH + h1 + r0) * LL + s0 * 8);
    const unsigned long long gaK1 = (unsigned long long)(const char*)
        (KT + ((size_t)b * HH + h1 + 16 + r0) * LL + s0 * 8);
    const unsigned int ldsbase = (unsigned int)(size_t)(void*)&ldsA[0][0];
    const unsigned int laK0 = ldsbase + r0 * 256 + s0 * 16;
    const unsigned int laK1 = laK0 + 16 * 256;

    const __bf16* brow0 = VT + ((size_t)b * HH + h2w + col) * LL;
    const __bf16* brow1 = brow0 + (size_t)16 * LL;

    AccF acc[2][2];
#pragma unroll
    for (int i = 0; i < 2; ++i)
#pragma unroll
        for (int j = 0; j < 2; ++j)
#pragma unroll
            for (int v = 0; v < 8; ++v) acc[i][j].e[v] = 0.f;

#define ISSUE_CHUNK(ch)                                                        \
    do {                                                                       \
        const unsigned int      bo  = ((ch) & 1) * 8192;                       \
        const unsigned long long go = (unsigned long long)(ch) * 256;          \
        unsigned long long ga = gaK0 + go;                                     \
        unsigned int       la = laK0 + bo;                                     \
        asm volatile("global_load_async_to_lds_b128 %0, %1, off"               \
                     :: "v"(la), "v"(ga) : "memory");                          \
        ga = gaK1 + go;                                                        \
        la = laK1 + bo;                                                        \
        asm volatile("global_load_async_to_lds_b128 %0, %1, off"               \
                     :: "v"(la), "v"(ga) : "memory");                          \
    } while (0)

    ISSUE_CHUNK(0);

    for (int ch = 0; ch < LL / 128; ++ch) {
        asm volatile("s_wait_asynccnt 0x0" ::: "memory");
        __syncthreads();                       // all waves' chunks landed
        if (ch + 1 < LL / 128) ISSUE_CHUNK(ch + 1);

        const char* abase = (const char*)&ldsA[ch & 1][0];
        const int l0 = ch * 128;
        __builtin_prefetch(brow0 + l0 + 128, 0, 1);   // global_prefetch_b8
        __builtin_prefetch(brow1 + l0 + 128, 0, 1);

#pragma unroll
        for (int kk = 0; kk < 4; ++kk) {
            FragBF a0, a1, b0, b1;
            const int ao = (kk * 32 + half * 16) * 2;  // byte offset in chunk
            a0.u2[0] = *(const uint4*)(abase + col * 256 + ao);
            a0.u2[1] = *(const uint4*)(abase + col * 256 + ao + 16);
            a1.u2[0] = *(const uint4*)(abase + (16 + col) * 256 + ao);
            a1.u2[1] = *(const uint4*)(abase + (16 + col) * 256 + ao + 16);
            const __bf16* bp0 = brow0 + l0 + kk * 32 + half * 16;
            b0.u2[0] = *(const uint4*)(bp0);
            b0.u2[1] = *(const uint4*)(bp0 + 8);
            const __bf16* bp1 = brow1 + l0 + kk * 32 + half * 16;
            b1.u2[0] = *(const uint4*)(bp1);
            b1.u2[1] = *(const uint4*)(bp1 + 8);
            acc[0][0].v = WMMA_BF16(a0.v, b0.v, acc[0][0].v);
            acc[0][1].v = WMMA_BF16(a0.v, b1.v, acc[0][1].v);
            acc[1][0].v = WMMA_BF16(a1.v, b0.v, acc[1][0].v);
            acc[1][1].v = WMMA_BF16(a1.v, b1.v, acc[1][1].v);
        }
        __syncthreads();                       // safe to overwrite this buffer
    }
#undef ISSUE_CHUNK

    // out = prev_cache + acc (f32)
#pragma unroll
    for (int i = 0; i < 2; ++i)
#pragma unroll
        for (int j = 0; j < 2; ++j)
#pragma unroll
            for (int v = 0; v < 8; ++v) {
                const int m = 16 * i + v + 8 * half;
                const size_t o =
                    ((size_t)b * HH + h1 + m) * HH + h2w + 16 * j + col;
                out[o] = C[o] + acc[i][j].e[v];
            }
}

// ---------------------------------------------------------------------------
extern "C" void kernel_launch(void* const* d_in, const int* in_sizes, int n_in,
                              void* d_out, int out_size, void* d_ws, size_t ws_size,
                              hipStream_t stream) {
    const float* hs = (const float*)d_in[0];   // [B, L, R]
    const float* C  = (const float*)d_in[1];   // [B, H, H]
    const float* kw = (const float*)d_in[2];   // [R, H]
    const float* kb = (const float*)d_in[3];   // [H]
    const float* vw = (const float*)d_in[4];   // [R, H]
    const float* vb = (const float*)d_in[5];   // [H]
    float* out = (float*)d_out;                // [B, H, H]

    // Workspace layout (bf16), ~138.5 MB total:
    //   WkT [H,R] | WvT [H,R] | CT [B,H,H] | KT [B,H,L] | VT [B,H,L]
    __bf16* WkT = (__bf16*)d_ws;
    __bf16* WvT = WkT + (size_t)HH * RR;
    __bf16* CT  = WvT + (size_t)HH * RR;
    __bf16* KT  = CT  + (size_t)BB * HH * HH;
    __bf16* VT  = KT  + (size_t)BB * HH * LL;

    prep_weights<<<(HH * RR + 255) / 256, 256, 0, stream>>>(kw, vw, WkT, WvT);
    prep_cache<<<(BB * HH * HH) / 256, 256, 0, stream>>>(C, CT);
    k_proj<<<BB * (LL / 16), 256, 0, stream>>>(hs, kb, vb, WkT, WvT, CT, KT, VT);
    k_outer<<<32 * BB, 256, 0, stream>>>(KT, VT, C, out);
}
